// HCGAttention_16587163698036
// MI455X (gfx1250) — matrix operand order repaired and checked by reading, not compiled, not run on hardware
//
#include <hip/hip_runtime.h>
#include <hip/hip_bf16.h>

// Problem constants (from reference): S=2048, B=8, H=512, DK=DV=64
#define SEQ   2048
#define BATCH 8
#define HID   512
#define DKV   64

static constexpr float NEG_INF_F = -1.0e9f;

typedef __attribute__((ext_vector_type(16))) __bf16 v16bf;
typedef __attribute__((ext_vector_type(8)))  float  v8f;
typedef __attribute__((ext_vector_type(2)))  __bf16 bf16x2;
typedef __attribute__((ext_vector_type(2)))  float  f32x2;
typedef __attribute__((ext_vector_type(4)))  unsigned int v4u;
typedef __attribute__((ext_vector_type(8)))  int    v8i;
typedef __attribute__((ext_vector_type(4)))  int    v4i;

// A-matrix (16x32 bf16) K index for VGPR v (0..7), lane-half h (0/1).
__device__ __forceinline__ int a_kidx(int v, int h) {
  return 2 * v + 16 * h;
}

// ---------------------------------------------------------------------------
// Tensor Data Mover: 2-D tile load Global->LDS.  D# packing per ISA 8.3-8.6:
//  group0: count=1 | lds_addr | global_addr[56:0] | type=2
//  group1: data_size=1(2B) | tensor_dim0/1 | tile_dim0/1 | dim0_stride
//  groups 2/3: zero (2-D tensor, tile_dim2..4 unused)
// dims/strides/tiles in elements (bf16).
// ---------------------------------------------------------------------------
__device__ __forceinline__ void tdm_load_2d_bf16(unsigned ldsAddr, const void* gptr,
                                                 unsigned dim0, unsigned dim1,
                                                 unsigned stride0,
                                                 unsigned tile0, unsigned tile1) {
  unsigned long long ga = (unsigned long long)(size_t)gptr;
  v4u g0;
  g0[0] = 1u;                                                // count=1, user desc
  g0[1] = ldsAddr;                                           // LDS byte address
  g0[2] = (unsigned)(ga & 0xffffffffu);                      // global_addr[31:0]
  g0[3] = (unsigned)((ga >> 32) & 0x1ffffffu) | (2u << 30);  // [56:32] | type=2
  v8i g1;
  g1[0] = (int)(1u << 16);                       // data_size=1 (2 bytes)
  g1[1] = (int)(dim0 << 16);                     // tensor_dim0[15:0] @ bits63:48
  g1[2] = (int)((dim0 >> 16) | (dim1 << 16));    // dim0 hi | tensor_dim1 lo
  g1[3] = (int)((dim1 >> 16) | (tile0 << 16));   // dim1 hi | tile_dim0
  g1[4] = (int)(tile1 & 0xffffu);                // tile_dim1 (tile_dim2=0)
  g1[5] = (int)stride0;                          // tensor_dim0_stride[31:0]
  g1[6] = 0;                                     // stride0 hi | dim1_stride lo
  g1[7] = 0;
  v4i gz = {0, 0, 0, 0};
#if __clang_major__ >= 23
  v8i gz8 = {0, 0, 0, 0, 0, 0, 0, 0};
  __builtin_amdgcn_tensor_load_to_lds(g0, g1, gz, gz, gz8, 0);
#else
  __builtin_amdgcn_tensor_load_to_lds(g0, g1, gz, gz, 0);
#endif
}

// ---------------------------------------------------------------------------
// Stage 1: fused projections  Q=(qWq^T+bq)/8,  K=(mask(k)Wk^T+bk),  V=vWv^T+bv
// One wave = 64 output rows (4 M-tiles) x DK=64, so each W k-slice fragment is
// reused by 4 WMMA (quarters the W re-read traffic vs 16-row waves).
// Q,K stored bf16 row-major (b,s,d); V stored transposed (b,d,s).
// ---------------------------------------------------------------------------
__global__ void __launch_bounds__(128)
proj_kernel(const float* __restrict__ qin, const float* __restrict__ kin,
            const float* __restrict__ vin,
            const float* __restrict__ Wq, const float* __restrict__ bq,
            const float* __restrict__ Wk, const float* __restrict__ bk,
            const float* __restrict__ Wv, const float* __restrict__ bv,
            const unsigned char* __restrict__ kpm,   // key_padding_mask (B,S)
            __bf16* __restrict__ Qws, __bf16* __restrict__ Kws,
            __bf16* __restrict__ VTws)
{
  const int lane = threadIdx.x & 31;
  const int wave = threadIdx.x >> 5;
  const int gw   = blockIdx.x * 4 + wave;    // 0..767
  const int mat  = gw >> 8;                  // 0=Q 1=K 2=V  (256 supertiles each)
  const int t64  = gw & 255;                 // 64-row supertile
  const int b    = t64 >> 5;                 // SEQ/64 = 32 supertiles per batch
  const int s64  = t64 & 31;
  const int half = lane >> 4;
  const int ncol = lane & 15;                // N index (B/C layout) and A row

  const float* X    = (mat == 0) ? qin : (mat == 1) ? kin : vin;
  const float* W    = (mat == 0) ? Wq  : (mat == 1) ? Wk  : Wv;
  const float* bias = (mat == 0) ? bq  : (mat == 1) ? bk  : bv;

  const float* xrow[4];
  bool maskedA[4];
#pragma unroll
  for (int mt = 0; mt < 4; ++mt) {
    const int srowA = s64 * 64 + mt * 16 + ncol;   // A row (M = lane&15)
    maskedA[mt] = (mat == 1) && (kpm[b * SEQ + srowA] != 0);
    xrow[mt]    = X + ((size_t)srowA * BATCH + b) * HID;
  }

  v8f acc[4][4] = {};   // [M-tile][N-tile]

  for (int h0 = 0; h0 < HID; h0 += 32) {
    // B fragments: W^T (h, d) 32x16 per N-tile, loaded once per k-step
    v16bf bm[4];
#pragma unroll
    for (int nt = 0; nt < 4; ++nt) {
      const int d = nt * 16 + ncol;
#pragma unroll
      for (int v = 0; v < 8; ++v) {
        const int kk = 2 * v + 16 * half;
        f32x2 w = *(const f32x2*)(W + (size_t)d * HID + h0 + kk);
        bm[nt][2 * v]     = (__bf16)w.x;
        bm[nt][2 * v + 1] = (__bf16)w.y;
      }
    }
#pragma unroll
    for (int mt = 0; mt < 4; ++mt) {
      v16bf a;
#pragma unroll
      for (int v = 0; v < 8; ++v) {
        const int kk = a_kidx(v, half);
        f32x2 x = *(const f32x2*)(xrow[mt] + h0 + kk);
        a[2 * v]     = (__bf16)(maskedA[mt] ? NEG_INF_F : x.x);
        a[2 * v + 1] = (__bf16)(maskedA[mt] ? NEG_INF_F : x.y);
      }
#pragma unroll
      for (int nt = 0; nt < 4; ++nt)
        acc[mt][nt] = __builtin_amdgcn_wmma_f32_16x16x32_bf16(
            false, a, false, bm[nt], (short)0, acc[mt][nt], false, false);
    }
  }

  // Epilogue: bias (+ fold 1/sqrt(64) into Q), store bf16 per C layout
  const int mbase = 8 * half;                // C row = v + mbase
#pragma unroll
  for (int mt = 0; mt < 4; ++mt) {
#pragma unroll
    for (int nt = 0; nt < 4; ++nt) {
      const int d  = nt * 16 + ncol;
      const float bs = bias[d];
#pragma unroll
      for (int v = 0; v < 8; ++v) {
        const int s = s64 * 64 + mt * 16 + v + mbase;
        float val = acc[mt][nt][v] + bs;
        if (mat == 0) {
          Qws[((size_t)b * SEQ + s) * DKV + d] = (__bf16)(val * 0.125f);
        } else if (mat == 1) {
          Kws[((size_t)b * SEQ + s) * DKV + d] = (__bf16)val;
        } else {
          VTws[((size_t)b * DKV + d) * SEQ + s] = (__bf16)val;   // transposed
        }
      }
    }
  }
}

// ---------------------------------------------------------------------------
// Stage 2: flash attention. One wave = 16 query rows x DV=64; 4 waves/block
// share batch b. K/V^T tiles (32 keys) are DMA'd once per block into LDS by
// the TDM (wave 0), next tile's DMA overlapping current tile's softmax+WMMA.
// ---------------------------------------------------------------------------
__global__ void __launch_bounds__(128)
attn_kernel(const __bf16* __restrict__ Qws, const __bf16* __restrict__ Kws,
            const __bf16* __restrict__ VTws,
            const unsigned char* __restrict__ amask,  // attn_mask (S,B)
            float* __restrict__ out)                  // (S,B,DV) f32
{
  __shared__ __bf16 klds[32 * 64];          // 32 keys x 64 d   (4KB)
  __shared__ __bf16 vlds[64 * 32];          // 64 dv  x 32 keys (4KB)
  __shared__ __bf16 plds[4][16 * 32];       // P staging, 1KB/wave

  const int lane  = threadIdx.x & 31;
  const int wave  = threadIdx.x >> 5;
  const int gw    = blockIdx.x * 4 + wave;  // 0..1023
  const int b     = gw >> 7;                // same for all 4 waves in block
  const int qtile = gw & 127;
  const int half  = lane >> 4;
  const int ncol  = lane & 15;
  const int mbase = 8 * half;
  __bf16* pl = &plds[wave][0];

  const __bf16* Kb  = Kws  + (size_t)b * SEQ * DKV;
  const __bf16* VTb = VTws + (size_t)b * DKV * SEQ;
  const unsigned kldsA = (unsigned)(size_t)(void*)klds;
  const unsigned vldsA = (unsigned)(size_t)(void*)vlds;

  // Prologue DMA for kb = 0 (issued once per block by wave 0)
  if (wave == 0) {
    tdm_load_2d_bf16(kldsA, Kb,  DKV, SEQ, DKV, DKV, 32);  // K: 32 rows x 64 d
    tdm_load_2d_bf16(vldsA, VTb, SEQ, DKV, SEQ, 32, DKV);  // VT: 64 rows x 32 s
  }

  // Q A-fragments (pre-scaled by 1/8): two k-steps over d = 0..63
  const __bf16* qrow = Qws + ((size_t)b * SEQ + qtile * 16 + ncol) * DKV;
  v16bf qa[2];
#pragma unroll
  for (int ks = 0; ks < 2; ++ks)
#pragma unroll
    for (int v = 0; v < 8; ++v) {
      const int d = 32 * ks + a_kidx(v, half);
      bf16x2 p = *(const bf16x2*)(qrow + d);
      qa[ks][2 * v] = p.x;  qa[ks][2 * v + 1] = p.y;
    }

  // Row mask flags (attn_mask masks whole query rows) for C rows v+mbase
  bool rowmasked[8];
#pragma unroll
  for (int v = 0; v < 8; ++v) {
    const int s = qtile * 16 + v + mbase;
    rowmasked[v] = amask[(size_t)s * BATCH + b] != 0;
  }

  float m[8], l[8];
  v8f acc[4] = {};
#pragma unroll
  for (int v = 0; v < 8; ++v) { m[v] = -3.0e38f; l[v] = 0.0f; }

  for (int kb = 0; kb < SEQ; kb += 32) {
    // --- wait for this tile's DMA, pull fragments out of LDS --------------
    if (wave == 0) __builtin_amdgcn_s_wait_tensorcnt(0);
    __syncthreads();

    v16bf kf[2][2];                           // [N-tile][k-step] B: (d x key)
#pragma unroll
    for (int nt = 0; nt < 2; ++nt) {
      const int key = nt * 16 + ncol;
#pragma unroll
      for (int ks = 0; ks < 2; ++ks)
#pragma unroll
        for (int v = 0; v < 8; ++v) {
          const int d = 32 * ks + 2 * v + 16 * half;
          bf16x2 p = *(const bf16x2*)(&klds[key * DKV + d]);
          kf[nt][ks][2 * v] = p.x;  kf[nt][ks][2 * v + 1] = p.y;
        }
    }
    v16bf vf[4];                              // B: (key x dv) 32x16 per N-tile
#pragma unroll
    for (int nt = 0; nt < 4; ++nt) {
      const int dv = nt * 16 + ncol;
#pragma unroll
      for (int v = 0; v < 8; ++v) {
        const int kk = 2 * v + 16 * half;
        bf16x2 p = *(const bf16x2*)(&vlds[dv * 32 + kk]);
        vf[nt][2 * v] = p.x;  vf[nt][2 * v + 1] = p.y;
      }
    }
    __syncthreads();
    // Kick next tile's DMA; it overlaps the softmax + WMMA below.
    if (wave == 0 && kb + 32 < SEQ) {
      tdm_load_2d_bf16(kldsA, Kb + (size_t)(kb + 32) * DKV, DKV, SEQ, DKV, DKV, 32);
      tdm_load_2d_bf16(vldsA, VTb + (kb + 32),              SEQ, DKV, SEQ, 32, DKV);
    }

    // --- scores S = Q K^T (already scaled): two 16x16 N-tiles -------------
    v8f sc[2];
#pragma unroll
    for (int nt = 0; nt < 2; ++nt) {
      v8f st = {};
#pragma unroll
      for (int ks = 0; ks < 2; ++ks)
        st = __builtin_amdgcn_wmma_f32_16x16x32_bf16(
            false, qa[ks], false, kf[nt][ks], (short)0, st, false, false);
      sc[nt] = st;
    }

    // --- masked-row replacement + online softmax --------------------------
    float s0[8], s1[8], t[8];
#pragma unroll
    for (int v = 0; v < 8; ++v) {
      s0[v] = rowmasked[v] ? NEG_INF_F : sc[0][v];
      s1[v] = rowmasked[v] ? NEG_INF_F : sc[1][v];
      t[v]  = fmaxf(s0[v], s1[v]);
    }
#pragma unroll
    for (int sh = 1; sh <= 8; sh <<= 1)       // row max over 16 lanes
#pragma unroll
      for (int v = 0; v < 8; ++v) t[v] = fmaxf(t[v], __shfl_xor(t[v], sh, 32));

    float p0[8], p1[8], alpha[8], rr[8];
#pragma unroll
    for (int v = 0; v < 8; ++v) {
      const float mn = fmaxf(m[v], t[v]);
      alpha[v] = __expf(m[v] - mn);
      p0[v]    = __expf(s0[v] - mn);
      p1[v]    = __expf(s1[v] - mn);
      rr[v]    = p0[v] + p1[v];
      m[v]     = mn;
    }
#pragma unroll
    for (int sh = 1; sh <= 8; sh <<= 1)       // row sum over 16 lanes
#pragma unroll
      for (int v = 0; v < 8; ++v) rr[v] += __shfl_xor(rr[v], sh, 32);
#pragma unroll
    for (int v = 0; v < 8; ++v) l[v] = l[v] * alpha[v] + rr[v];
#pragma unroll
    for (int nt = 0; nt < 4; ++nt)
#pragma unroll
      for (int v = 0; v < 8; ++v) acc[nt][v] *= alpha[v];

    // --- restage P (C layout) -> A layout via LDS -------------------------
#pragma unroll
    for (int v = 0; v < 8; ++v) {
      pl[(v + mbase) * 32 + ncol]      = (__bf16)p0[v];
      pl[(v + mbase) * 32 + 16 + ncol] = (__bf16)p1[v];
    }
    __syncthreads();
    v16bf pa;                                 // A: 16 q-rows x 32 keys
#pragma unroll
    for (int v = 0; v < 8; ++v) {
      const int kk = a_kidx(v, half);
      bf16x2 p = *(const bf16x2*)(pl + ncol * 32 + kk);
      pa[2 * v] = p.x;  pa[2 * v + 1] = p.y;
    }
    __syncthreads();

    // --- O += P V : 4 output N-tiles of 16 --------------------------------
#pragma unroll
    for (int nt = 0; nt < 4; ++nt)
      acc[nt] = __builtin_amdgcn_wmma_f32_16x16x32_bf16(
          false, pa, false, vf[nt], (short)0, acc[nt], false, false);
  }

  // Epilogue: normalize and store (S,B,DV) f32
#pragma unroll
  for (int nt = 0; nt < 4; ++nt) {
    const int dv = nt * 16 + ncol;
#pragma unroll
    for (int v = 0; v < 8; ++v) {
      const int s = qtile * 16 + v + mbase;
      out[((size_t)s * BATCH + b) * DKV + dv] = acc[nt][v] / l[v];
    }
  }
}

// ---------------------------------------------------------------------------
extern "C" void kernel_launch(void* const* d_in, const int* in_sizes, int n_in,
                              void* d_out, int out_size, void* d_ws, size_t ws_size,
                              hipStream_t stream) {
  const float* qin = (const float*)d_in[0];
  const float* kin = (const float*)d_in[1];
  const float* vin = (const float*)d_in[2];
  const float* Wq  = (const float*)d_in[3];
  const float* bq  = (const float*)d_in[4];
  const float* Wk  = (const float*)d_in[5];
  const float* bk  = (const float*)d_in[6];
  const float* Wv  = (const float*)d_in[7];
  const float* bv  = (const float*)d_in[8];
  const unsigned char* amask = (const unsigned char*)d_in[9];   // (S,B) bool
  const unsigned char* kpm   = (const unsigned char*)d_in[10];  // (B,S) bool

  // Workspace: Q (2MB) | K (2MB) | V^T (2MB), bf16
  const size_t mat_elems = (size_t)BATCH * SEQ * DKV;           // 1,048,576
  __bf16* Qws  = (__bf16*)d_ws;
  __bf16* Kws  = Qws + mat_elems;
  __bf16* VTws = Kws + mat_elems;

  // Stage 1: 3 matrices x 256 64-row supertiles, 4 waves/block -> 192 blocks
  proj_kernel<<<192, 128, 0, stream>>>(qin, kin, vin, Wq, bq, Wk, bk, Wv, bv,
                                       kpm, Qws, Kws, VTws);
  // Stage 2: 1024 (b, q-tile) waves, 4 waves/block -> 256 blocks
  attn_kernel<<<256, 128, 0, stream>>>(Qws, Kws, VTws, amask, (float*)d_out);
}
